// SlidingDecoderBlock_36043365548101
// MI455X (gfx1250) — matrix operand order repaired and checked by reading, not compiled, not run on hardware
//
#include <hip/hip_runtime.h>
#include <hip/hip_bf16.h>

typedef __attribute__((ext_vector_type(16))) __bf16 v16bf;
typedef __attribute__((ext_vector_type(8)))  __bf16 v8bf;
typedef __attribute__((ext_vector_type(8)))  float  v8f;

#define DEV static __device__ __forceinline__

// ---------------------------------------------------------------------------
// WMMA fragment helpers (layouts per CDNA5 ISA 7.12.2, wave32)
// A 16x32 bf16: lane l holds row (l&15); kb=(l<16)?0:8; elems0..7 -> K=kb..kb+7,
//               elems8..15 -> K=16+kb..16+kb+7.  p must point at A[row]+kb.
// B 32x16 bf16: lane l holds col (l&15); kb=(l<16)?0:8; elems0..7 -> K=kb..kb+7,
//               elems8..15 -> K=16+kb..16+kb+7.
//               p must point at Bt[col]+kb (Bt is B transposed, row-major NxK).
// C/D 16x16 f32: lane l holds col (l&15); VGPR i -> row i + ((l<16)?0:8).
// ---------------------------------------------------------------------------
DEV v16bf ld_frag_a(const __bf16* p) {
  v8bf lo = *(const v8bf*)p;
  v8bf hi = *(const v8bf*)(p + 16);
  return __builtin_shufflevector(lo, hi, 0,1,2,3,4,5,6,7,8,9,10,11,12,13,14,15);
}
DEV v16bf ld_frag_b(const __bf16* p) {
  v8bf lo = *(const v8bf*)p;
  v8bf hi = *(const v8bf*)(p + 16);
  return __builtin_shufflevector(lo, hi, 0,1,2,3,4,5,6,7,8,9,10,11,12,13,14,15);
}
DEV v8f wmma_bf16(v16bf a, v16bf b, v8f c) {
  return __builtin_amdgcn_wmma_f32_16x16x32_bf16(false, a, false, b, (short)0, c,
                                                 false, false);
}

// async copy 16B from global to LDS (gfx1250 ASYNCcnt path)
DEV void async_g2l_b128(uint32_t lds_off, const void* gaddr) {
  asm volatile("global_load_async_to_lds_b128 %0, %1, off"
               :: "v"(lds_off), "v"((uint64_t)(uintptr_t)gaddr) : "memory");
}

// ---------------------------------------------------------------------------
// Elementwise / prep kernels
// ---------------------------------------------------------------------------
__global__ __launch_bounds__(256)
void wtrans_kernel(const float* __restrict__ W, __bf16* __restrict__ Wt,
                   int K, int N) {
  size_t idx = (size_t)blockIdx.x * 256 + threadIdx.x;
  if (idx >= (size_t)K * N) return;
  int k = (int)(idx / N);
  int n = (int)(idx % N);
  Wt[(size_t)n * K + k] = (__bf16)W[idx];
}

__global__ __launch_bounds__(256)
void cvt_bf16_kernel(const float* __restrict__ in, __bf16* __restrict__ out,
                     size_t n) {
  size_t idx = (size_t)blockIdx.x * 256 + threadIdx.x;
  if (idx < n) out[idx] = (__bf16)in[idx];
}

// one block (256 thr) per row of length Dn
__global__ __launch_bounds__(256)
void rmsnorm_kernel(const float* __restrict__ x, const float* __restrict__ w,
                    __bf16* __restrict__ out, int Dn) {
  const int row = blockIdx.x;
  const float* xr = x + (size_t)row * Dn;
  float ss = 0.0f;
  for (int d = threadIdx.x; d < Dn; d += 256) { float v = xr[d]; ss += v * v; }
#pragma unroll
  for (int off = 16; off > 0; off >>= 1) ss += __shfl_xor(ss, off, 32);
  __shared__ float red[8];
  if ((threadIdx.x & 31) == 0) red[threadIdx.x >> 5] = ss;
  __syncthreads();
  float tot = 0.0f;
#pragma unroll
  for (int i = 0; i < 8; ++i) tot += red[i];
  const float inv = rsqrtf(tot / (float)Dn + 1e-6f);
  __bf16* orow = out + (size_t)row * Dn;
  for (int d = threadIdx.x; d < Dn; d += 256)
    orow[d] = (__bf16)(xr[d] * inv * w[d]);
}

// ---------------------------------------------------------------------------
// Shared GEMM epilogues
// ---------------------------------------------------------------------------
enum { EPI_F32 = 0, EPI_BF16 = 1, EPI_F32_ADD = 2, EPI_BF16_VT = 3, EPI_SILU_BF16 = 4 };

template <int EPI>
DEV void epi_store(void* outp, const void* aux, int m, int n, int Nt, float v,
                   int vtH, int vtHD, int vtS) {
  if constexpr (EPI == EPI_F32) {
    ((float*)outp)[(size_t)m * Nt + n] = v;
  } else if constexpr (EPI == EPI_BF16) {
    ((__bf16*)outp)[(size_t)m * Nt + n] = (__bf16)v;
  } else if constexpr (EPI == EPI_F32_ADD) {
    ((float*)outp)[(size_t)m * Nt + n] =
        v + ((const float*)aux)[(size_t)m * Nt + n];
  } else if constexpr (EPI == EPI_BF16_VT) {
    // store transposed per head: out[b][h][d][s]  (m=(b,s), n=(h,d))
    const int bb = m / vtS, s = m % vtS;
    const int hh = n / vtHD, d = n % vtHD;
    ((__bf16*)outp)[((size_t)(bb * vtH + hh) * vtHD + d) * vtS + s] = (__bf16)v;
  } else {  // EPI_SILU_BF16: out = silu(aux) * acc
    const float g  = ((const float*)aux)[(size_t)m * Nt + n];
    const float sg = g / (1.0f + __expf(-g));
    ((__bf16*)outp)[(size_t)m * Nt + n] = (__bf16)(sg * v);
  }
}

// ---------------------------------------------------------------------------
// Simple GEMM (one 16x16 C tile per wave) for small-N shapes.
// ---------------------------------------------------------------------------
template <int EPI>
__global__ __launch_bounds__(256)
void gemm_simple_kernel(const __bf16* __restrict__ A, const __bf16* __restrict__ Bt,
                        void* __restrict__ outp, const void* __restrict__ aux,
                        int Mt, int Nt, int Kt) {
  const int lane   = threadIdx.x & 31;
  const int wv     = threadIdx.x >> 5;
  const int tilesN = Nt >> 4;
  const int tile   = blockIdx.x * 8 + wv;
  if (tile >= (Mt >> 4) * tilesN) return;
  const int row0 = (tile / tilesN) << 4;
  const int col0 = (tile % tilesN) << 4;
  const int hihalf = (lane & 16) ? 1 : 0;
  const __bf16* ap = A  + (size_t)(row0 + (lane & 15)) * Kt + (hihalf ? 8 : 0);
  const __bf16* bp = Bt + (size_t)(col0 + (lane & 15)) * Kt + (hihalf ? 8 : 0);
  v8f acc = {};
  for (int k0 = 0; k0 < Kt; k0 += 32)
    acc = wmma_bf16(ld_frag_a(ap + k0), ld_frag_b(bp + k0), acc);
  const int n    = col0 + (lane & 15);
  const int roff = hihalf ? 8 : 0;
#pragma unroll
  for (int i = 0; i < 8; ++i)
    epi_store<EPI>(outp, aux, row0 + roff + i, n, Nt, acc[i], 0, 0, 0);
}

// ---------------------------------------------------------------------------
// Block-tiled GEMM: 256 threads = 8 waves own a 128x64 C tile.
// B tile (64 cols x 32 k) is double-buffered in LDS via
// global_load_async_to_lds_b128 (ASYNCcnt), shared by all 8 waves; each wave
// computes 16 rows x 64 cols so the A fragment is reused across 4 WMMAs.
// LDS B layout: col-major rows, padded stride 40 elems -> conflict-free b128.
// Grid must be exactly (Mt/128)*(Nt/64) blocks (no divergent barriers).
// ---------------------------------------------------------------------------
#define BSTRIDE 40  // elems; 80B per col, 16B aligned, bank-conflict-free

template <int EPI>
__global__ __launch_bounds__(256)
void gemm_tiled_kernel(const __bf16* __restrict__ A, const __bf16* __restrict__ Bt,
                       void* __restrict__ outp, const void* __restrict__ aux,
                       int Mt, int Nt, int Kt, int vtH, int vtHD, int vtS) {
  __shared__ __attribute__((aligned(16))) __bf16 Bsh[2][64 * BSTRIDE];
  const int tid  = threadIdx.x;
  const int lane = tid & 31;
  const int wv   = tid >> 5;
  const int tilesM = Mt >> 7;
  const int tm = blockIdx.x % tilesM;
  const int tn = blockIdx.x / tilesM;
  const int row0 = (tm << 7) + (wv << 4);
  const int col0 = tn << 6;

  const int hihalf = (lane & 16) ? 1 : 0;
  const int rA   = lane & 15;
  const int kbA  = hihalf ? 8 : 0;
  const int kbB  = hihalf ? 8 : 0;
  const int colq = lane & 15;
  const __bf16* ap = A + (size_t)(row0 + rA) * Kt + kbA;

  // async staging map: thread -> (B col, 16B chunk of 32-k row)
  const int bcol  = tid >> 2;
  const int bchnk = tid & 3;
  const __bf16* bgp = Bt + (size_t)(col0 + bcol) * Kt + bchnk * 8;
  const uint32_t ldsColOff = (uint32_t)(bcol * (BSTRIDE * 2) + bchnk * 16);
  const uint32_t base0 = (uint32_t)(uintptr_t)(&Bsh[0][0]);
  const uint32_t base1 = (uint32_t)(uintptr_t)(&Bsh[1][0]);

  async_g2l_b128(base0 + ldsColOff, bgp);  // prologue: k-tile 0 -> buf 0

  v8f acc[4] = {};
  const int nk = Kt >> 5;
  for (int i = 0; i < nk; ++i) {
    asm volatile("s_wait_asynccnt 0x0" ::: "memory");
    __syncthreads();  // this wave's + all waves' tile-i bytes are in LDS
    if (i + 1 < nk) {
      const uint32_t dst = ((i + 1) & 1) ? base1 : base0;
      async_g2l_b128(dst + ldsColOff, bgp + (size_t)(i + 1) * 32);
    }
    const __bf16* bb = &Bsh[i & 1][0] + (size_t)colq * BSTRIDE + kbB;
    const v16bf a = ld_frag_a(ap + i * 32);
    acc[0] = wmma_bf16(a, ld_frag_b(bb), acc[0]);
    acc[1] = wmma_bf16(a, ld_frag_b(bb + 16 * BSTRIDE), acc[1]);
    acc[2] = wmma_bf16(a, ld_frag_b(bb + 32 * BSTRIDE), acc[2]);
    acc[3] = wmma_bf16(a, ld_frag_b(bb + 48 * BSTRIDE), acc[3]);
  }

  const int roff = hihalf ? 8 : 0;
#pragma unroll
  for (int nb = 0; nb < 4; ++nb) {
    const int n = col0 + nb * 16 + colq;
#pragma unroll
    for (int i = 0; i < 8; ++i)
      epi_store<EPI>(outp, aux, row0 + roff + i, n, Nt, acc[nb][i],
                     vtH, vtHD, vtS);
  }
}

// ---------------------------------------------------------------------------
// Causal self-attention, flash style. One wave = 16-query tile of one (b,h).
// q,k: (B,S,H,64) bf16.  vT: (B,H,64,S) bf16.  o: (B,S,H,64) bf16.
// ---------------------------------------------------------------------------
__global__ __launch_bounds__(256)
void self_attn_kernel(const __bf16* __restrict__ q, const __bf16* __restrict__ k,
                      const __bf16* __restrict__ vT, __bf16* __restrict__ o,
                      int Bn, int Hn, int Sn) {
  __shared__ __attribute__((aligned(16))) __bf16 Pst[8][16 * 32];
  const int lane  = threadIdx.x & 31;
  const int wv    = threadIdx.x >> 5;
  const int qtPer = Sn >> 4;
  const int t     = blockIdx.x * 8 + wv;
  if (t >= Bn * Hn * qtPer) return;
  const int b  = t / (Hn * qtPer);
  const int rr = t % (Hn * qtPer);
  const int h  = rr / qtPer;
  const int q0 = (rr % qtPer) << 4;

  const int hihalf = (lane & 16) ? 1 : 0;
  const int rA   = lane & 15;
  const int kbA  = hihalf ? 8 : 0;
  const int kbB  = hihalf ? 8 : 0;
  const int colq = lane & 15;
  const int roff = hihalf ? 8 : 0;
  const int ldq  = Hn * 64;

  const size_t headOff = ((size_t)b * Sn * Hn + h) * 64;
  const __bf16* qp  = q + headOff + (size_t)(q0 + rA) * ldq + kbA;
  const v16bf aq0 = ld_frag_a(qp);        // q dims  0..31
  const v16bf aq1 = ld_frag_a(qp + 32);   // q dims 32..63
  const __bf16* kp0 = k + headOff + kbB;  // + key*ldq
  const __bf16* vb  = vT + ((size_t)(b * Hn + h) * 64) * Sn;  // + d*Sn + key

  float mrow[8], lrow[8];
  v8f oa0 = {}, oa1 = {}, oa2 = {}, oa3 = {};
#pragma unroll
  for (int i = 0; i < 8; ++i) { mrow[i] = -1e30f; lrow[i] = 0.0f; }

  __bf16* P = &Pst[wv][0];

  for (int j0 = 0; j0 <= q0 + 15; j0 += 32) {
    // ---- scores: S = Q (16x64) @ K^T, two 16-key tiles ----
    v8f s0 = {}, s1 = {};
    {
      const __bf16* kr0 = kp0 + (size_t)(j0 + colq) * ldq;
      s0 = wmma_bf16(aq0, ld_frag_b(kr0), s0);
      s0 = wmma_bf16(aq1, ld_frag_b(kr0 + 32), s0);
      const __bf16* kr1 = kp0 + (size_t)(j0 + 16 + colq) * ldq;
      s1 = wmma_bf16(aq0, ld_frag_b(kr1), s1);
      s1 = wmma_bf16(aq1, ld_frag_b(kr1 + 32), s1);
    }
    // ---- scale, causal mask, online softmax (16-lane row reductions) ----
#pragma unroll
    for (int i = 0; i < 8; ++i) {
      const int qi = q0 + roff + i;
      float x0 = (j0 + colq      <= qi) ? s0[i] * 0.125f : -1e30f;
      float x1 = (j0 + 16 + colq <= qi) ? s1[i] * 0.125f : -1e30f;
      float mx = fmaxf(x0, x1);
#pragma unroll
      for (int off = 1; off < 16; off <<= 1) mx = fmaxf(mx, __shfl_xor(mx, off, 32));
      const float nm = fmaxf(mrow[i], mx);
      const float sc = __expf(mrow[i] - nm);
      const float p0 = __expf(x0 - nm);
      const float p1 = __expf(x1 - nm);
      float ps = p0 + p1;
#pragma unroll
      for (int off = 1; off < 16; off <<= 1) ps += __shfl_xor(ps, off, 32);
      lrow[i] = lrow[i] * sc + ps;
      mrow[i] = nm;
      oa0[i] *= sc; oa1[i] *= sc; oa2[i] *= sc; oa3[i] *= sc;
      P[(roff + i) * 32 + colq]      = (__bf16)p0;
      P[(roff + i) * 32 + 16 + colq] = (__bf16)p1;
    }
    // C-layout -> A-layout transpose through per-wave LDS staging
    asm volatile("s_wait_dscnt 0x0" ::: "memory");
    const v16bf pa = ld_frag_a(P + rA * 32 + kbA);
    asm volatile("" ::: "memory");
    // ---- O += P (16x32) @ V (32 x 64), four 16-dim column blocks ----
    oa0 = wmma_bf16(pa, ld_frag_b(vb + (size_t)( 0 + colq) * Sn + j0 + kbB), oa0);
    oa1 = wmma_bf16(pa, ld_frag_b(vb + (size_t)(16 + colq) * Sn + j0 + kbB), oa1);
    oa2 = wmma_bf16(pa, ld_frag_b(vb + (size_t)(32 + colq) * Sn + j0 + kbB), oa2);
    oa3 = wmma_bf16(pa, ld_frag_b(vb + (size_t)(48 + colq) * Sn + j0 + kbB), oa3);
  }
#pragma unroll
  for (int i = 0; i < 8; ++i) {
    const float inv = 1.0f / lrow[i];
    const size_t ob = ((size_t)(b * Sn + (q0 + roff + i)) * Hn + h) * 64;
    o[ob +  0 + colq] = (__bf16)(oa0[i] * inv);
    o[ob + 16 + colq] = (__bf16)(oa1[i] * inv);
    o[ob + 32 + colq] = (__bf16)(oa2[i] * inv);
    o[ob + 48 + colq] = (__bf16)(oa3[i] * inv);
  }
}

// ---------------------------------------------------------------------------
// Windowed cross-attention: each query attends to keys [seg-8, seg] (<=9 keys).
// One thread per (b,s,h). qc:(B,S,H,64) kc,vc:(B,M,H,64) oc:(B,S,H,64) bf16.
// ---------------------------------------------------------------------------
__global__ __launch_bounds__(256)
void cross_attn_kernel(const __bf16* __restrict__ qc, const __bf16* __restrict__ kc,
                       const __bf16* __restrict__ vc, const int* __restrict__ seg,
                       __bf16* __restrict__ oc, int Bn, int Sn, int Mn, int Hn) {
  const int idx = blockIdx.x * 256 + threadIdx.x;
  if (idx >= Bn * Sn * Hn) return;
  const int h  = idx % Hn;
  const int bs = idx / Hn;
  const int s  = bs % Sn;
  const int b  = bs / Sn;
  const int sg = seg[b * Sn + s];
  int lo = sg - 8; if (lo < 0) lo = 0;
  const int cnt = sg - lo + 1;  // <= 9

  const __bf16* qp = qc + ((size_t)(b * Sn + s) * Hn + h) * 64;
  float sc[9];
  float mx = -1e30f;
  for (int j = 0; j < cnt; ++j) {
    const __bf16* kp = kc + ((size_t)(b * Mn + lo + j) * Hn + h) * 64;
    float d = 0.0f;
#pragma unroll
    for (int e = 0; e < 64; ++e) d += (float)qp[e] * (float)kp[e];
    sc[j] = d * 0.125f;
    mx = fmaxf(mx, sc[j]);
  }
  float l = 0.0f;
  for (int j = 0; j < cnt; ++j) { sc[j] = __expf(sc[j] - mx); l += sc[j]; }
  const float inv = 1.0f / l;
  float acc[64];
#pragma unroll
  for (int e = 0; e < 64; ++e) acc[e] = 0.0f;
  for (int j = 0; j < cnt; ++j) {
    const __bf16* vp = vc + ((size_t)(b * Mn + lo + j) * Hn + h) * 64;
    const float p = sc[j];
#pragma unroll
    for (int e = 0; e < 64; ++e) acc[e] += p * (float)vp[e];
  }
  __bf16* op = oc + ((size_t)(b * Sn + s) * Hn + h) * 64;
#pragma unroll
  for (int e = 0; e < 64; ++e) op[e] = (__bf16)(acc[e] * inv);
}

// ---------------------------------------------------------------------------
// Host orchestration
// ---------------------------------------------------------------------------
extern "C" void kernel_launch(void* const* d_in, const int* in_sizes, int n_in,
                              void* d_out, int out_size, void* d_ws, size_t ws_size,
                              hipStream_t stream) {
  (void)in_sizes; (void)n_in; (void)out_size; (void)ws_size;
  const int Bb = 2, Ss = 2048, Mm = 512, Dd = 1024, Hh = 16, HDd = 64;
  const int QCc = 32, KVCc = 64, DF = 4096;
  const int BS = Bb * Ss;   // 4096 token rows
  const int BM = Bb * Mm;   // 1024 memory rows

  const float* x_in    = (const float*)d_in[0];
  const float* memory  = (const float*)d_in[1];
  const int*   seg     = (const int*)d_in[2];
  const float* norm1_w = (const float*)d_in[3];
  const float* W_dq    = (const float*)d_in[4];
  const float* W_uq    = (const float*)d_in[5];
  const float* W_dkv   = (const float*)d_in[6];
  const float* W_uk    = (const float*)d_in[7];
  const float* W_uv    = (const float*)d_in[8];
  const float* W_osl   = (const float*)d_in[9];
  const float* norm2_w = (const float*)d_in[10];
  const float* W_qc    = (const float*)d_in[11];
  const float* W_kc    = (const float*)d_in[12];
  const float* W_vc    = (const float*)d_in[13];
  const float* W_oc    = (const float*)d_in[14];
  const float* norm3_w = (const float*)d_in[15];
  const float* W_gate  = (const float*)d_in[16];
  const float* W_up    = (const float*)d_in[17];
  const float* W_down  = (const float*)d_in[18];

  char* wsp = (char*)d_ws;
  size_t off = 0;
  auto alloc = [&](size_t bytes) -> void* {
    off = (off + 255) & ~(size_t)255;
    void* p = wsp + off;
    off += bytes;
    return p;
  };
  const size_t BF = sizeof(__bf16);

  float*  xw1  = (float*)alloc((size_t)BS * Dd * 4);
  float*  xw2  = (float*)alloc((size_t)BS * Dd * 4);
  __bf16* hb   = (__bf16*)alloc((size_t)BS * Dd * BF);
  __bf16* memb = (__bf16*)alloc((size_t)BM * Dd * BF);
  __bf16* Wt_dq   = (__bf16*)alloc((size_t)QCc * Dd * BF);
  __bf16* Wt_uq   = (__bf16*)alloc((size_t)Dd * QCc * BF);
  __bf16* Wt_dkv  = (__bf16*)alloc((size_t)KVCc * Dd * BF);
  __bf16* Wt_uk   = (__bf16*)alloc((size_t)Dd * KVCc * BF);
  __bf16* Wt_uv   = (__bf16*)alloc((size_t)Dd * KVCc * BF);
  __bf16* Wt_osl  = (__bf16*)alloc((size_t)Dd * Dd * BF);
  __bf16* Wt_qc   = (__bf16*)alloc((size_t)Dd * Dd * BF);
  __bf16* Wt_kc   = (__bf16*)alloc((size_t)Dd * Dd * BF);
  __bf16* Wt_vc   = (__bf16*)alloc((size_t)Dd * Dd * BF);
  __bf16* Wt_oc   = (__bf16*)alloc((size_t)Dd * Dd * BF);
  __bf16* Wt_gate = (__bf16*)alloc((size_t)DF * Dd * BF);
  __bf16* Wt_up   = (__bf16*)alloc((size_t)DF * Dd * BF);
  __bf16* Wt_down = (__bf16*)alloc((size_t)Dd * DF * BF);
  __bf16* t1  = (__bf16*)alloc((size_t)BS * QCc * BF);
  __bf16* ckv = (__bf16*)alloc((size_t)BS * KVCc * BF);
  __bf16* qb  = (__bf16*)alloc((size_t)BS * Dd * BF);
  __bf16* kb  = (__bf16*)alloc((size_t)BS * Dd * BF);
  __bf16* vTb = (__bf16*)alloc((size_t)BS * Dd * BF);
  __bf16* ob  = (__bf16*)alloc((size_t)BS * Dd * BF);
  __bf16* qcb = (__bf16*)alloc((size_t)BS * Dd * BF);
  __bf16* kcb = (__bf16*)alloc((size_t)BM * Dd * BF);
  __bf16* vcb = (__bf16*)alloc((size_t)BM * Dd * BF);
  __bf16* ocb = (__bf16*)alloc((size_t)BS * Dd * BF);
  float*  gb  = (float*)alloc((size_t)BS * DF * 4);
  __bf16* act = (__bf16*)alloc((size_t)BS * DF * BF);

  auto cdiv = [](size_t a, size_t b) { return (unsigned)((a + b - 1) / b); };
  auto tr = [&](const float* W, __bf16* Wt, int K, int N) {
    wtrans_kernel<<<cdiv((size_t)K * N, 256), 256, 0, stream>>>(W, Wt, K, N);
  };
  auto tblk = [&](int M, int N) { return (unsigned)((M / 128) * (N / 64)); };

  // weight prep (bf16, transposed to NxK)
  tr(W_dq, Wt_dq, Dd, QCc);     tr(W_uq, Wt_uq, QCc, Dd);
  tr(W_dkv, Wt_dkv, Dd, KVCc);  tr(W_uk, Wt_uk, KVCc, Dd);
  tr(W_uv, Wt_uv, KVCc, Dd);    tr(W_osl, Wt_osl, Dd, Dd);
  tr(W_qc, Wt_qc, Dd, Dd);      tr(W_kc, Wt_kc, Dd, Dd);
  tr(W_vc, Wt_vc, Dd, Dd);      tr(W_oc, Wt_oc, Dd, Dd);
  tr(W_gate, Wt_gate, Dd, DF);  tr(W_up, Wt_up, Dd, DF);
  tr(W_down, Wt_down, DF, Dd);
  cvt_bf16_kernel<<<cdiv((size_t)BM * Dd, 256), 256, 0, stream>>>(
      memory, memb, (size_t)BM * Dd);

  // ---- stage 1: self attention (MLA-style low-rank QKV) ----
  rmsnorm_kernel<<<BS, 256, 0, stream>>>(x_in, norm1_w, hb, Dd);
  gemm_simple_kernel<EPI_BF16><<<cdiv((size_t)(BS / 16) * (QCc / 16), 8), 256, 0,
                                 stream>>>(hb, Wt_dq, t1, nullptr, BS, QCc, Dd);
  gemm_tiled_kernel<EPI_BF16><<<tblk(BS, Dd), 256, 0, stream>>>(
      t1, Wt_uq, qb, nullptr, BS, Dd, QCc, 0, 0, 0);
  gemm_tiled_kernel<EPI_BF16><<<tblk(BS, KVCc), 256, 0, stream>>>(
      hb, Wt_dkv, ckv, nullptr, BS, KVCc, Dd, 0, 0, 0);
  gemm_tiled_kernel<EPI_BF16><<<tblk(BS, Dd), 256, 0, stream>>>(
      ckv, Wt_uk, kb, nullptr, BS, Dd, KVCc, 0, 0, 0);
  gemm_tiled_kernel<EPI_BF16_VT><<<tblk(BS, Dd), 256, 0, stream>>>(
      ckv, Wt_uv, vTb, nullptr, BS, Dd, KVCc, Hh, HDd, Ss);
  self_attn_kernel<<<cdiv((size_t)Bb * Hh * (Ss / 16), 8), 256, 0, stream>>>(
      qb, kb, vTb, ob, Bb, Hh, Ss);
  gemm_tiled_kernel<EPI_F32_ADD><<<tblk(BS, Dd), 256, 0, stream>>>(
      ob, Wt_osl, xw1, x_in, BS, Dd, Dd, 0, 0, 0);

  // ---- stage 2: windowed cross attention over memory ----
  rmsnorm_kernel<<<BS, 256, 0, stream>>>(xw1, norm2_w, hb, Dd);
  gemm_tiled_kernel<EPI_BF16><<<tblk(BS, Dd), 256, 0, stream>>>(
      hb, Wt_qc, qcb, nullptr, BS, Dd, Dd, 0, 0, 0);
  gemm_tiled_kernel<EPI_BF16><<<tblk(BM, Dd), 256, 0, stream>>>(
      memb, Wt_kc, kcb, nullptr, BM, Dd, Dd, 0, 0, 0);
  gemm_tiled_kernel<EPI_BF16><<<tblk(BM, Dd), 256, 0, stream>>>(
      memb, Wt_vc, vcb, nullptr, BM, Dd, Dd, 0, 0, 0);
  cross_attn_kernel<<<cdiv((size_t)Bb * Ss * Hh, 256), 256, 0, stream>>>(
      qcb, kcb, vcb, seg, ocb, Bb, Ss, Mm, Hh);
  gemm_tiled_kernel<EPI_F32_ADD><<<tblk(BS, Dd), 256, 0, stream>>>(
      ocb, Wt_oc, xw2, xw1, BS, Dd, Dd, 0, 0, 0);

  // ---- stage 3: SwiGLU MLP ----
  rmsnorm_kernel<<<BS, 256, 0, stream>>>(xw2, norm3_w, hb, Dd);
  gemm_tiled_kernel<EPI_F32><<<tblk(BS, DF), 256, 0, stream>>>(
      hb, Wt_gate, gb, nullptr, BS, DF, Dd, 0, 0, 0);
  gemm_tiled_kernel<EPI_SILU_BF16><<<tblk(BS, DF), 256, 0, stream>>>(
      hb, Wt_up, act, gb, BS, DF, Dd, 0, 0, 0);
  gemm_tiled_kernel<EPI_F32_ADD><<<tblk(BS, Dd), 256, 0, stream>>>(
      act, Wt_down, (float*)d_out, xw2, BS, Dd, DF, 0, 0, 0);
}
